// LinearBase_2628519985669
// MI455X (gfx1250) — 8-bit matrix operand order repaired and checked by reading, compiled, not run on hardware
//
#include <hip/hip_runtime.h>


typedef __attribute__((ext_vector_type(8)))  int   v8i;
typedef __attribute__((ext_vector_type(8)))  float v8f;

#define FP8_MAX 448.0f

#if __has_builtin(__builtin_amdgcn_cvt_pk_fp8_f32)
__device__ __forceinline__ int pack_fp8x4(float a, float b, float c, float d) {
  int r = __builtin_amdgcn_cvt_pk_fp8_f32(a, b, 0, false);
  r     = __builtin_amdgcn_cvt_pk_fp8_f32(c, d, r, true);
  return r;
}
#else
__device__ __forceinline__ unsigned fp8_byte(float f) {
  unsigned u    = __float_as_uint(f);
  unsigned sign = (u >> 24) & 0x80u;
  unsigned ab   = u & 0x7fffffffu;
  if (ab >= 0x7f800000u) return sign | 0x7fu;
  if (__uint_as_float(ab) > 448.0f) return sign | 0x7eu;
  unsigned lsb = (ab >> 20) & 1u;
  ab += 0x7ffffu + lsb;
  int e8 = (int)(ab >> 23) - 127 + 7;
  unsigned mant = (ab >> 20) & 7u;
  unsigned v = (e8 >= 1) ? (((unsigned)e8 << 3) | mant) : 0u;
  if (v > 0x7eu) v = 0x7eu;
  return sign | v;
}
__device__ __forceinline__ int pack_fp8x4(float a, float b, float c, float d) {
  return (int)(fp8_byte(a) | (fp8_byte(b) << 8) | (fp8_byte(c) << 16) |
               (fp8_byte(d) << 24));
}
#endif

__device__ __forceinline__ void async_cp_b128(unsigned lds_off, unsigned voff,
                                              unsigned long long sbase) {
  asm volatile("global_load_async_to_lds_b128 %0, %1, %2"
               :: "v"(lds_off), "v"(voff), "s"(sbase) : "memory");
}
__device__ __forceinline__ void wait_async_le8() {
  asm volatile("s_wait_asynccnt 0x8" ::: "memory");
}
__device__ __forceinline__ void wait_async_le0() {
  asm volatile("s_wait_asynccnt 0x0" ::: "memory");
}

__global__ void quant_rows_kernel(const float* __restrict__ x,
                                  unsigned char* __restrict__ xq,
                                  float* __restrict__ xsinv, int K) {
  const int wave = threadIdx.x >> 5;
  const int lane = threadIdx.x & 31;
  const float* row = x + (size_t)wave * K;

  float amax = 0.0f;
  for (int k = lane * 4; k < K; k += 128) {
    float4 v = *(const float4*)(row + k);
    amax = fmaxf(amax, fmaxf(fmaxf(fabsf(v.x), fabsf(v.y)),
                             fmaxf(fabsf(v.z), fabsf(v.w))));
  }
  #pragma unroll
  for (int m = 16; m >= 1; m >>= 1)
    amax = fmaxf(amax, __shfl_xor(amax, m, 32));

  const float scale = (amax > 0.0f) ? (FP8_MAX / amax) : 1.0f;
  const float inv   = (amax > 0.0f) ? (amax / FP8_MAX) : 1.0f;
  if (lane == 0) xsinv[wave] = inv;

  unsigned char* qrow = xq + (size_t)wave * K;
  for (int k = lane * 4; k < K; k += 128) {
    float4 v = *(const float4*)(row + k);
    *(int*)(qrow + k) =
        pack_fp8x4(v.x * scale, v.y * scale, v.z * scale, v.w * scale);
  }
}

template <int CN, int CK>
__global__ void __launch_bounds__(256)
fp8_gemm_async_kernel(const unsigned char* __restrict__ xq,
                      const float* __restrict__ xsinv,
                      const float* __restrict__ W,
                      const float* __restrict__ wsinv,
                      float* __restrict__ partial,
                      int M, int rtK, int rtN, int kPerSplit) {
  const int N = (CN > 0) ? CN : rtN;
  const int K = (CK > 0) ? CK : rtK;
  constexpr int TK = 64;
  constexpr int TN = 128;
  __shared__ float tile[2][TK * TN];

  const int tid  = threadIdx.x;
  const int wave = tid >> 5;
  const int lane = tid & 31;
  const int hi   = lane >> 4;
  const int lq   = lane & 15;

  const int c0     = blockIdx.x * TN;
  const int kBase  = blockIdx.y * kPerSplit;
  const int nTiles = kPerSplit / TK;

  const unsigned pvoff0 = (unsigned)wave * (unsigned)N * 4u + (unsigned)lane * 16u;
  const unsigned voffStep = (unsigned)N * 32u;
  const unsigned ldsBase0 = (unsigned)(size_t)(&tile[0][0]);
  const unsigned plds0 = ldsBase0 + (unsigned)wave * (TN * 4u) + (unsigned)lane * 16u;

  auto issue_tile = [&](int buf, int k0) {
    const unsigned long long g =
        (unsigned long long)(size_t)(W + (size_t)k0 * N + c0);
    const unsigned ldsb = plds0 + (unsigned)buf * (unsigned)(TK * TN * 4);
    unsigned vo = pvoff0;
    #pragma unroll
    for (int r = 0; r < 8; ++r) {
      async_cp_b128(ldsb + (unsigned)r * (TN * 4u * 8u), vo, g);
      vo += voffStep;
    }
  };

  v8f acc = {};
  const unsigned char* ab0 = xq + (size_t)lq * K + kBase + hi * 8;
  const int lc = wave * 16 + lq;

  issue_tile(0, kBase);
  for (int t = 0; t < nTiles; ++t) {
    if (t + 1 < nTiles) {
      issue_tile((t + 1) & 1, kBase + (t + 1) * TK);
      wait_async_le8();
    } else {
      wait_async_le0();
    }
    __syncthreads();

    v8i a;
    const unsigned char* ab = ab0 + t * TK;
    #pragma unroll
    for (int p = 0; p < 4; ++p) {
      int2 d = *(const int2*)(ab + p * 16);
      a[p * 2 + 0] = d.x;
      a[p * 2 + 1] = d.y;
    }

    v8i b;
    const float* cb = &tile[t & 1][0] + hi * 8 * TN + lc;
    #pragma unroll
    for (int j = 0; j < 8; ++j) {
      const int kr = (j >> 1) * 16 + (j & 1) * 4;
      b[j] = pack_fp8x4(cb[(kr + 0) * TN], cb[(kr + 1) * TN],
                        cb[(kr + 2) * TN], cb[(kr + 3) * TN]);
    }

    acc = __builtin_amdgcn_wmma_f32_16x16x64_fp8_fp8(a, b, (short)0, acc,
                                                     false, false);
    __syncthreads();
  }

  const int col = c0 + lc;
  const float wi = wsinv[col];
  #pragma unroll
  for (int v = 0; v < 8; ++v) {
    const int m = v + hi * 8;
    if (m < M)
      partial[((size_t)blockIdx.y * M + m) * N + col] = acc[v] * xsinv[m] * wi;
  }
}

__global__ void reduce_bias_kernel(const float* __restrict__ partial,
                                   const float* __restrict__ bias,
                                   float* __restrict__ out,
                                   int S, int M, int N) {
  const int i = blockIdx.x * blockDim.x + threadIdx.x;
  if (i >= M * N) return;
  const int n = i % N;
  float s = bias[n];
  for (int t = 0; t < S; ++t) s += partial[(size_t)t * M * N + i];
  out[i] = s;
}

extern "C" void kernel_launch(void* const* d_in, const int* in_sizes, int n_in,
                              void* d_out, int out_size, void* d_ws, size_t ws_size,
                              hipStream_t stream) {
  (void)n_in; (void)out_size; (void)ws_size;
  const float* x     = (const float*)d_in[0];
  const float* w     = (const float*)d_in[1];
  const float* wsinv = (const float*)d_in[2];
  const float* bias  = (const float*)d_in[3];
  float* out = (float*)d_out;

  const int N = in_sizes[2];
  const int K = in_sizes[1] / N;
  const int M = in_sizes[0] / K;

  const int S = (K % 256 == 0) ? 4 : 1;
  const int kPerSplit = K / S;

  unsigned char* xq      = (unsigned char*)d_ws;
  float*         xsinv   = (float*)(xq + (size_t)16 * K);
  float*         partial = (float*)(xq + (size_t)16 * K + 64);

  hipLaunchKernelGGL(quant_rows_kernel, dim3(1), dim3(M * 32), 0, stream,
                     x, xq, xsinv, K);

  dim3 grid(N / 128, S);
  if (N == 8192 && K == 8192) {
    hipLaunchKernelGGL((fp8_gemm_async_kernel<8192, 8192>), grid, dim3(256), 0,
                       stream, xq, xsinv, w, wsinv, partial, M, K, N, kPerSplit);
  } else {
    hipLaunchKernelGGL((fp8_gemm_async_kernel<0, 0>), grid, dim3(256), 0,
                       stream, xq, xsinv, w, wsinv, partial, M, K, N, kPerSplit);
  }

  const int total = M * N;
  hipLaunchKernelGGL(reduce_bias_kernel, dim3((total + 255) / 256), dim3(256),
                     0, stream, partial, bias, out, S, M, N);
}
